// CIN_69569880261393
// MI455X (gfx1250) — matrix operand order repaired and checked by reading, not compiled, not run on hardware
//
#include <hip/hip_runtime.h>

typedef _Float16 half_t;
typedef __attribute__((ext_vector_type(16))) _Float16 v16h;
typedef __attribute__((ext_vector_type(8)))  float    v8f;

#define B_    512
#define F0_   40
#define D_    32
#define NOUT  200
#define MT    13   // ceil(200/16) M-tiles

// ---------------------------------------------------------------- convert f32 -> f16
__global__ __launch_bounds__(256)
void cin_cvt_f32_f16(const float* __restrict__ src, half_t* __restrict__ dst, int n) {
  int i = blockIdx.x * 256 + threadIdx.x;
  if (i < n) dst[i] = (half_t)src[i];
}

// ---------------------------------------------------------------- pack W [K,200] f32
// into WMMA A-operand layout: wp[(t*nk + q)*32 + lane][16 halves]
// lane<16 : row M = t*16+lane,       K halves = {0..7, 16..23} of the 32-chunk
// lane>=16: row M = t*16+(lane-16),  K halves = {8..15, 24..31}
__global__ __launch_bounds__(256)
void cin_pack_w(const float* __restrict__ W, half_t* __restrict__ wp, int nk) {
  int idx = blockIdx.x * 256 + threadIdx.x;
  int total = MT * nk * 32;
  if (idx >= total) return;
  int lane = idx & 31;
  int tq   = idx >> 5;
  int q    = tq % nk;
  int t    = tq / nk;
  int n    = t * 16 + (lane & 15);
  int kb   = q * 32 + ((lane & 16) >> 1);   // +8 for the hi lane group
  half_t* dst = wp + (size_t)idx * 16;
#pragma unroll
  for (int m = 0; m < 16; ++m) {
    int k = kb + (m < 8 ? m : m + 8);
    float v = (n < NOUT) ? W[(size_t)k * NOUT + n] : 0.0f;
    dst[m] = (half_t)v;
  }
}

// ---------------------------------------------------------------- one CIN layer
// Per block: batch b.  C[208,32] = Wpackᵀ x Z, Z[k,d] = x0[k/Fi,d]*h[k%Fi,d]
// K is processed 64 at a time (two 32-chunks) -> 8 WMMA per wave per barrier pair.
__global__ __launch_bounds__(256)
void cin_layer(const half_t* __restrict__ x0h,    // [B, F0, 32]
               const half_t* __restrict__ hprev,  // [B, Fi, 32]
               const half_t* __restrict__ wpack,  // packed A operands
               half_t* __restrict__ hnext,        // [B, 200, 32] f16 (or null)
               float*  __restrict__ out,          // [B, 600] f32
               int Fi, int nkc, int out_off)      // nkc = K/64 iterations
{
  __shared__ half_t s_x0[F0_ * D_];      // 2.5 KB
  __shared__ half_t s_h [NOUT * D_];     // 12.5 KB
  __shared__ v16h   s_zb[2][2][32];      // 4 KB: [k-chunk][d-tile][lane] B operands

  const int b    = blockIdx.x;
  const int tid  = threadIdx.x;
  const int lane = tid & 31;
  const int wv   = __builtin_amdgcn_readfirstlane(tid >> 5);   // scalar wave id

  // stage x0 and h for this batch into LDS
  for (int i = tid; i < F0_ * D_; i += 256)
    s_x0[i] = x0h[(size_t)b * F0_ * D_ + i];
  const int hn = Fi * D_;
  for (int i = tid; i < hn; i += 256)
    s_h[i] = hprev[(size_t)b * hn + i];

  // per-thread Z-fill slots (8 halves each), flattened over [kc][dt][lane][16]
  half_t* s_z = (half_t*)s_zb;
  int zi[8], zj[8], zd[8];
#pragma unroll
  for (int u = 0; u < 8; ++u) {
    int flat = tid * 8 + u;               // 0..2047
    int kc   = flat >> 10;                // which 32-K chunk
    int zl   = (flat >> 4) & 31;          // lane slot inside tile
    int m    = flat & 15;                 // half index within lane
    zd[u] = (((flat >> 9) & 1) << 4) + (zl & 15);   // embed dim d
    int kl = kc * 32 + ((zl & 16) >> 1) + (m < 8 ? m : m + 8);   // k_local in 0..63
    zi[u] = 0;
    zj[u] = kl;
  }

  const int t0   = wv;
  const int t1   = wv + 8;
  const int has1 = (t1 < MT);

  v8f a00 = {}, a01 = {}, a10 = {}, a11 = {};

  const int nk = nkc * 2;                 // #32-chunks (wpack granularity)
  const half_t* wp0 = wpack + (size_t)t0 * nk * 32 * 16;
  const half_t* wp1 = wpack + (size_t)t1 * nk * 32 * 16;

  __syncthreads();

  for (int it = 0; it < nkc; ++it) {
    // build Z super-chunk (64 K x 32 D) in B-operand layout
#pragma unroll
    for (int u = 0; u < 8; ++u)
      s_z[tid * 8 + u] = s_x0[zi[u] * D_ + zd[u]] * s_h[zj[u] * D_ + zd[u]];
    __syncthreads();

    const size_t qb = (size_t)(it * 2) * 32 * 16;   // chunk base in wpack, in halves
#pragma unroll
    for (int kc = 0; kc < 2; ++kc) {
      v16h b0 = s_zb[kc][0][lane];
      v16h b1 = s_zb[kc][1][lane];
      v16h A0 = *(const v16h*)(wp0 + qb + ((size_t)kc * 32 + lane) * 16);
      a00 = __builtin_amdgcn_wmma_f32_16x16x32_f16(false, A0, false, b0, (short)0, a00, false, false);
      a01 = __builtin_amdgcn_wmma_f32_16x16x32_f16(false, A0, false, b1, (short)0, a01, false, false);
      if (has1) {
        v16h A1 = *(const v16h*)(wp1 + qb + ((size_t)kc * 32 + lane) * 16);
        a10 = __builtin_amdgcn_wmma_f32_16x16x32_f16(false, A1, false, b0, (short)0, a10, false, false);
        a11 = __builtin_amdgcn_wmma_f32_16x16x32_f16(false, A1, false, b1, (short)0, a11, false, false);
      }
    }

    if (it + 1 < nkc) {
      // near-cache prefetch of next iteration's A operand streams
      const size_t qn = (size_t)((it + 1) * 2) * 32 * 16 + (size_t)lane * 16;
      __builtin_prefetch(wp0 + qn, 0, 3);
      __builtin_prefetch(wp1 + qn, 0, 3);
      // advance (i,j): k += 64 may cross up to two rows when Fi == 40
#pragma unroll
      for (int u = 0; u < 8; ++u) {
        zj[u] += 64;
        if (zj[u] >= Fi) { zj[u] -= Fi; zi[u] += 1; }
        if (zj[u] >= Fi) { zj[u] -= Fi; zi[u] += 1; }
      }
    }
    __syncthreads();
  }

  // epilogue: C layout -> f16 h_next + sum over d into out
  const int g = lane >> 4;     // lane group: rows r / r+8
  const int c = lane & 15;     // column within d-tile
  auto flush = [&](int t, const v8f& X, const v8f& Y) {
#pragma unroll
    for (int r = 0; r < 8; ++r) {
      int n = t * 16 + g * 8 + r;
      float x = X[r], y = Y[r];
      if (hnext != nullptr && n < NOUT) {
        hnext[((size_t)b * NOUT + n) * D_ + c]      = (half_t)x;
        hnext[((size_t)b * NOUT + n) * D_ + 16 + c] = (half_t)y;
      }
      float s = x + y;                       // partial over both d-tiles
      s += __shfl_xor(s, 1, 32);             // butterfly within 16-lane group
      s += __shfl_xor(s, 2, 32);
      s += __shfl_xor(s, 4, 32);
      s += __shfl_xor(s, 8, 32);
      if (c == 0 && n < NOUT)
        out[(size_t)b * 600 + out_off + n] = s;
    }
  };
  flush(t0, a00, a01);
  if (has1) flush(t1, a10, a11);
}

// ---------------------------------------------------------------- host launcher
extern "C" void kernel_launch(void* const* d_in, const int* in_sizes, int n_in,
                              void* d_out, int out_size, void* d_ws, size_t ws_size,
                              hipStream_t stream) {
  const float* x0 = (const float*)d_in[0];   // [512,40,32]
  const float* W0 = (const float*)d_in[1];   // [1600,200]
  const float* W1 = (const float*)d_in[2];   // [8000,200]
  const float* W2 = (const float*)d_in[3];   // [8000,200]
  float* out = (float*)d_out;                // [512,600]

  char* ws = (char*)d_ws;
  size_t off = 0;
  auto take = [&](size_t bytes) -> void* {
    void* p = ws + off;
    off += (bytes + 255) & ~(size_t)255;
    return p;
  };
  const int nk0  = (F0_ * F0_) / 32;    // 50  32-chunks
  const int nk12 = (NOUT * F0_) / 32;   // 250 32-chunks

  half_t* x0h = (half_t*)take((size_t)B_ * F0_  * D_ * 2);
  half_t* h1  = (half_t*)take((size_t)B_ * NOUT * D_ * 2);
  half_t* h2  = (half_t*)take((size_t)B_ * NOUT * D_ * 2);
  half_t* wp0 = (half_t*)take((size_t)MT * nk0  * 32 * 16 * 2);
  half_t* wp1 = (half_t*)take((size_t)MT * nk12 * 32 * 16 * 2);
  half_t* wp2 = (half_t*)take((size_t)MT * nk12 * 32 * 16 * 2);

  {
    int n = B_ * F0_ * D_;
    cin_cvt_f32_f16<<<(n + 255) / 256, 256, 0, stream>>>(x0, x0h, n);
  }
  cin_pack_w<<<(MT * nk0  * 32 + 255) / 256, 256, 0, stream>>>(W0, wp0, nk0);
  cin_pack_w<<<(MT * nk12 * 32 + 255) / 256, 256, 0, stream>>>(W1, wp1, nk12);
  cin_pack_w<<<(MT * nk12 * 32 + 255) / 256, 256, 0, stream>>>(W2, wp2, nk12);

  cin_layer<<<B_, 256, 0, stream>>>(x0h, x0h, wp0, h1, out, F0_,  nk0  / 2, 0);
  cin_layer<<<B_, 256, 0, stream>>>(x0h, h1,  wp1, h2, out, NOUT, nk12 / 2, 200);
  cin_layer<<<B_, 256, 0, stream>>>(x0h, h2,  wp2, (half_t*)nullptr, out, NOUT, nk12 / 2, 400);
}
